// CRF_24464133718909
// MI455X (gfx1250) — compile-verified, matrix operand order checked by reading, not run on hardware
//
#include <hip/hip_runtime.h>

// ---------------------------------------------------------------------------
// CRF mean-field loop on gfx1250:  x <- x0 + w * SepGaussBlur9(softmax_C(x)),
// 5 iterations.  B=16, C=21, H=W=384.  Blur runs on the WMMA pipe:
//   T(32x16)  = Patch(32x32) x BandW(32x16)     (2x v_wmma_f32_16x16x32_f16)
//   Out(16x16)= BandH(16x32) x T(32x16)         (1x v_wmma_f32_16x16x32_f16)
// f16 operands (probs in [0,1]), f32 accumulation.
// ---------------------------------------------------------------------------

typedef __attribute__((ext_vector_type(16))) _Float16 v16h;
typedef __attribute__((ext_vector_type(8)))  _Float16 v8h;
typedef __attribute__((ext_vector_type(8)))  float    v8f;

namespace {
constexpr int Bsz = 16;
constexpr int Csz = 21;
constexpr int Hsz = 384;
constexpr int Wsz = 384;
constexpr int HW  = Hsz * Wsz;
constexpr int TILES_X = Wsz / 16;               // 24
constexpr int TILES_Y = Hsz / 16;               // 24
constexpr int TILES_PER_PLANE = TILES_X * TILES_Y;   // 576
constexpr int NPLANES = Bsz * Csz;              // 336
constexpr int NJOBS = NPLANES * TILES_PER_PLANE;     // 193536
constexpr int BLUR_BLOCKS = NJOBS / 8;          // 24192 (8 waves/block, 1 tile/wave)
constexpr int PIX = Bsz * HW;                   // 2359296
constexpr int PIX_BLOCKS = PIX / 256;           // 9216
}

// ----------------------------- layout transposes ---------------------------

__global__ __launch_bounds__(256) void nhwc_to_planar(const float* __restrict__ x,
                                                      float* __restrict__ xp) {
  int p = blockIdx.x * 256 + threadIdx.x;       // pixel over B*H*W (exact)
  int b = p / HW, hw = p - b * HW;
  size_t src = (size_t)p * Csz;
  size_t dst = (size_t)b * Csz * HW + hw;
#pragma unroll
  for (int c = 0; c < Csz; ++c) xp[dst + (size_t)c * HW] = x[src + c];
}

__global__ __launch_bounds__(256) void planar_to_nhwc(const float* __restrict__ xp,
                                                      float* __restrict__ out) {
  int p = blockIdx.x * 256 + threadIdx.x;
  int b = p / HW, hw = p - b * HW;
  size_t src = (size_t)b * Csz * HW + hw;
  size_t dst = (size_t)p * Csz;
#pragma unroll
  for (int c = 0; c < Csz; ++c) out[dst + c] = xp[src + (size_t)c * HW];
}

// ----------------------------- softmax over C ------------------------------

__global__ __launch_bounds__(256) void crf_softmax(const float* __restrict__ xc,
                                                   _Float16* __restrict__ p16) {
  int p = blockIdx.x * 256 + threadIdx.x;
  int b = p / HW, hw = p - b * HW;
  size_t base = (size_t)b * Csz * HW + hw;
  float v[Csz];
  float mx = -3.4e38f;
#pragma unroll
  for (int c = 0; c < Csz; ++c) { v[c] = xc[base + (size_t)c * HW]; mx = fmaxf(mx, v[c]); }
  float s = 0.f;
#pragma unroll
  for (int c = 0; c < Csz; ++c) { v[c] = __expf(v[c] - mx); s += v[c]; }
  float inv = 1.0f / s;
#pragma unroll
  for (int c = 0; c < Csz; ++c) p16[base + (size_t)c * HW] = (_Float16)(v[c] * inv);
}

// --------------------- WMMA separable blur + x0 update ---------------------
// One wave = one 16x16 output tile of one (b,c) plane.  8 waves / block.

__global__ __launch_bounds__(256) void crf_blur_wmma(
    const _Float16* __restrict__ p16, const float* __restrict__ x0p,
    float* __restrict__ xn,
    const float* __restrict__ spacing, const float* __restrict__ invtheta,
    const float* __restrict__ wptr) {
  // padded strides: 40 f16 (80B) and 24 f16 (48B) rows -> bank-spread, 16B aligned
  __shared__ __align__(16) _Float16 sPatch[8][32][40];
  __shared__ __align__(16) _Float16 sT[8][32][24];

  const int tid  = threadIdx.x;
  const int wv   = tid >> 5;
  const int lane = tid & 31;
  const int n    = lane & 15;
  const int half = lane >> 4;

  int job   = blockIdx.x * 8 + wv;              // grid sized exactly -> EXEC all-1s
  int plane = job / TILES_PER_PLANE;            // b*C + c
  int t     = job - plane * TILES_PER_PLANE;
  int th    = t / TILES_X, tw = t - th * TILES_X;
  int b     = plane / Csz;
  int h0 = th * 16, w0 = tw * 16;
  size_t pBase = (size_t)plane * HW;

  float sp_h = spacing[b * 2 + 0] * invtheta[0];   // taps along H
  float sp_w = spacing[b * 2 + 1] * invtheta[1];   // taps along W
  float wgt  = wptr[0];

  // ---- stage 32x32 f16 prob patch (rows h0-8..h0+23, cols w0-8..w0+23) ----
  {
    int gh  = h0 - 8 + lane;                     // one patch row per lane
    int gw0 = w0 - 8;
    bool rowIn = (gh >= 0) & (gh < Hsz);
    if (rowIn & (gw0 >= 0) & (gw0 + 32 <= Wsz)) {
      const uint4* src = (const uint4*)(p16 + pBase + (size_t)gh * Wsz + gw0); // 16B aligned
      uint4* dst = (uint4*)&sPatch[wv][lane][0];
      dst[0] = src[0]; dst[1] = src[1]; dst[2] = src[2]; dst[3] = src[3];
    } else {
#pragma unroll
      for (int j = 0; j < 32; ++j) {
        int gw = gw0 + j;
        _Float16 v = (_Float16)0.f;
        if (rowIn && gw >= 0 && gw < Wsz) v = p16[pBase + (size_t)gh * Wsz + gw];
        sPatch[wv][lane][j] = v;
      }
    }
  }
  asm volatile("s_wait_dscnt 0x0" ::: "memory");   // per-wave LDS region: intra-wave RAW only

  // K index held by element e of a v16h operand (A/B layouts, 16-bit, K=32):
  //   e<8 -> K = 8*half + e ; e>=8 -> K = 16 + 8*half + (e-8)
  // Horizontal band B-frag: Bh[k][n] = kw(k-n-8), zero outside |d|<=4 and at d=0.
  v16h bh;
#pragma unroll
  for (int e = 0; e < 16; ++e) {
    int k = (e < 8) ? (8 * half + e) : (16 + 8 * half + (e - 8));
    float dd = (float)(k - n - 8);
    float a  = dd * sp_w;
    float val = (dd != 0.f && dd >= -4.f && dd <= 4.f) ? __expf(-0.5f * a * a) : 0.f;
    bh[e] = (_Float16)val;
  }

  // A-frags: lane holds patch row (rowBase + n), two contiguous 16B chunks.
  const v8h* alo0 = (const v8h*)&sPatch[wv][n][8 * half];
  const v8h* ahi0 = (const v8h*)&sPatch[wv][n][16 + 8 * half];
  v16h a0 = __builtin_shufflevector(*alo0, *ahi0, 0,1,2,3,4,5,6,7,8,9,10,11,12,13,14,15);
  const v8h* alo1 = (const v8h*)&sPatch[wv][16 + n][8 * half];
  const v8h* ahi1 = (const v8h*)&sPatch[wv][16 + n][16 + 8 * half];
  v16h a1 = __builtin_shufflevector(*alo1, *ahi1, 0,1,2,3,4,5,6,7,8,9,10,11,12,13,14,15);

  v8f cz = {};
  v8f t0 = __builtin_amdgcn_wmma_f32_16x16x32_f16(false, a0, false, bh, (short)0, cz, false, false);
  v8f t1 = __builtin_amdgcn_wmma_f32_16x16x32_f16(false, a1, false, bh, (short)0, cz, false, false);

  // C-layout -> LDS (T rows 0..31 = patch rows), as f16 for the vertical pass
  const int mofs = half * 8;
#pragma unroll
  for (int r = 0; r < 8; ++r) {
    sT[wv][mofs + r][n]      = (_Float16)t0[r];
    sT[wv][16 + mofs + r][n] = (_Float16)t1[r];
  }
  asm volatile("s_wait_dscnt 0x0" ::: "memory");

  // Vertical: A = band Av[m][k] = kv(k-m-8) (computed in-regs), B = T from LDS.
  v16h av, bt;
#pragma unroll
  for (int e = 0; e < 16; ++e) {
    int k = (e < 8) ? (8 * half + e) : (16 + 8 * half + (e - 8));
    float dd = (float)(k - n - 8);
    float a  = dd * sp_h;
    av[e] = (_Float16)((dd != 0.f && dd >= -4.f && dd <= 4.f) ? __expf(-0.5f * a * a) : 0.f);
    bt[e] = sT[wv][k][n];
  }
  v8f acc = __builtin_amdgcn_wmma_f32_16x16x32_f16(false, av, false, bt, (short)0, cz, false, false);

  // x_next = x0 + w * blur ; C-layout: lane n, VGPR r -> (row mofs+r, col n)
#pragma unroll
  for (int r = 0; r < 8; ++r) {
    int gh2 = h0 + mofs + r;
    size_t idx = pBase + (size_t)gh2 * Wsz + (w0 + n);
    xn[idx] = fmaf(wgt, acc[r], x0p[idx]);
  }
}

// --------------------------------- driver ----------------------------------

extern "C" void kernel_launch(void* const* d_in, const int* in_sizes, int n_in,
                              void* d_out, int out_size, void* d_ws, size_t ws_size,
                              hipStream_t stream) {
  const float* x       = (const float*)d_in[0];   // (B,H,W,C) f32
  const float* spacing = (const float*)d_in[1];   // (B,2) f32
  const float* wgt     = (const float*)d_in[2];   // scalar
  const float* invth   = (const float*)d_in[3];   // (2,) f32

  float* out = (float*)d_out;                     // (B,H,W,C) f32; also ping-pong scratch

  size_t planeBytes = (size_t)Bsz * Csz * HW * sizeof(float);   // 198 MB
  float*    x0p  = (float*)d_ws;                                 // planar x0 (const)
  float*    bufA = (float*)((char*)d_ws + planeBytes);           // planar ping-pong A
  _Float16* p16  = (_Float16*)((char*)d_ws + 2 * planeBytes);    // f16 probs (99 MB)

  nhwc_to_planar<<<PIX_BLOCKS, 256, 0, stream>>>(x, x0p);

  const float* cur = x0p;
  float* dsts[5] = { bufA, out, bufA, out, bufA };
#pragma unroll
  for (int it = 0; it < 5; ++it) {
    crf_softmax<<<PIX_BLOCKS, 256, 0, stream>>>(cur, p16);
    crf_blur_wmma<<<BLUR_BLOCKS, 256, 0, stream>>>(p16, x0p, dsts[it], spacing, invth, wgt);
    cur = dsts[it];
  }

  planar_to_nhwc<<<PIX_BLOCKS, 256, 0, stream>>>(bufA, out);
}
